// Prediction_57939108823650
// MI455X (gfx1250) — matrix operand order repaired and checked by reading, not compiled, not run on hardware
//
#include <hip/hip_runtime.h>
#include <hip/hip_bf16.h>
#include <math.h>

// ---------------------------------------------------------------------------
// CDNA5 (gfx1250) GNN inference pipeline.
// All MLP matmuls run on v_wmma_f32_16x16x32_bf16 (bf16 A/B, f32 accum).
// MLP shapes are compile-time templates -> fully unrolled straight-line WMMA
// chains per kernel. Weights are swizzled into B-fragment order in LDS so a
// fragment load is 2x ds_load_b128; A-fragments use 128-bit LDS loads + cvt.
// 16-row tiles per wave32; segment-sums via global_atomic_add_f32
// (working set lives in the 192MB L2).
// ---------------------------------------------------------------------------

typedef __attribute__((ext_vector_type(16))) __bf16 bf16x16;
typedef __attribute__((ext_vector_type(8)))  float  f32x8;
typedef __attribute__((ext_vector_type(4)))  float  f32x4;

#define NWAVES 4   // 128-thread blocks: 4 wave32, one 16-row tile each

struct GatherSrc {
    const float* ptr;
    const int*   idx;       // null -> identity row index
    int rowStride;          // floats per source row
    int colOff;             // starting column within source row
    int width;              // columns copied
    int shift;              // row index >>= shift (for jnp.repeat(..., D) inputs)
};

struct MlpW { const float* W[5]; const float* B[5]; };

struct EpiP {
    float* o0; float* o1; float* o2;
    const float* a0; const float* a1;
    const int*   sidx;
    const float* tab; int ts, to;
    int n;
};

enum { EPI_ADD = 0, EPI_I2O_NC, EPI_I2O_SWI, EPI_STORE, EPI_SELECT,
       EPI_LUT, EPI_LUTIN, EPI_FSCALE };

__device__ __forceinline__ float sigmoidf_(float x) { return 1.0f / (1.0f + __expf(-x)); }

// Compile-time MLP shape descriptor: up to 5 layers, dims[0]=input width.
template <int NL, int D0, int D1, int D2, int D3, int D4, int D5>
struct Shape {
    static constexpr int nl = NL;
    static constexpr int dim(int i) { const int d[6] = {D0, D1, D2, D3, D4, D5}; return d[i]; }
    static constexpr int rup32(int x) { return (x + 31) & ~31; }
    static constexpr int rup16(int x) { return (x + 15) & ~15; }
    static constexpr int stride() {
        int s = 32;
        for (int l = 0; l < NL; ++l) { const int k = rup32(dim(l)); if (k > s) s = k; }
        return s;
    }
    static constexpr int woff(int l) {
        int o = 0;
        for (int i = 0; i < l; ++i) o += rup32(dim(i)) * rup16(dim(i + 1));
        return o;
    }
    static constexpr int boff(int l) {
        int o = 0;
        for (int i = 0; i < l; ++i) o += rup16(dim(i + 1));
        return o;
    }
    static constexpr size_t smem() {
        return (((size_t)woff(NL) * 2 + 15) & ~(size_t)15) + (size_t)boff(NL) * 4 +
               (size_t)NWAVES * 2 * 16 * stride() * 4;
    }
};

// A-fragment: 16x32 bf16 tile of the activation buffer (f32 in LDS -> bf16).
// Lane L: row m = L&15; lane half selects K chunk {0..7,16..23} vs {8..15,24..31}.
// Addresses are 32B-aligned -> 4x ds_load_b128 + v_cvt_pk_bf16_f32.
template <int STRIDE>
__device__ __forceinline__ bf16x16 afrag(const float* buf, int kb0, int lane) {
    const int m  = lane & 15;
    const int kb = kb0 + ((lane >> 4) << 3);
    const f32x4* p = (const f32x4*)(buf + m * STRIDE + kb);
    const f32x4 q0 = p[0];   // k = kb+0..3
    const f32x4 q1 = p[1];   // k = kb+4..7
    const f32x4 q2 = p[4];   // k = kb+16..19
    const f32x4 q3 = p[5];   // k = kb+20..23
    bf16x16 a;
#pragma unroll
    for (int i = 0; i < 4; ++i) {
        a[i]      = (__bf16)q0[i];
        a[i + 4]  = (__bf16)q1[i];
        a[i + 8]  = (__bf16)q2[i];
        a[i + 12] = (__bf16)q3[i];
    }
    return a;
}

// B-fragment: weights pre-swizzled in LDS as [ktile][ntile][lane][16 bf16],
// so a fragment is one 32-byte contiguous vector load per lane (2x ds_load_b128).
template <int NT>
__device__ __forceinline__ bf16x16 bfrag(const __bf16* w, int kt, int nt, int lane) {
    return *(const bf16x16*)(w + (((kt * NT + nt) * 32 + lane) << 4));
}

// Cooperative weight convert f32 -> bf16 into LDS in B-fragment order
// (zero-pad K and N), unrolled over layers.
template <class S, int L>
__device__ __forceinline__ void load_weights(const MlpW& wp, __bf16* wbase, float* bbase) {
    if constexpr (L < S::nl) {
        constexpr int K = S::dim(L), N = S::dim(L + 1);
        constexpr int Kp = S::rup32(K), N16 = S::rup16(N);
        constexpr int NT = N16 / 16;
        const float* W = wp.W[L];
        for (int idx = threadIdx.x; idx < Kp * N16; idx += blockDim.x) {
            const int i    = idx & 15;          // element within lane chunk
            const int lane = (idx >> 4) & 31;   // owning lane
            const int t    = idx >> 9;          // fragment tile = kt*NT + nt
            const int nt   = t % NT;
            const int kt   = t / NT;
            // B-layout: lane n-col = nt*16 + (lane&15); K = kt*32 + (laneHi?8:0) + (i&7) + (i>=8?16:0)
            const int k = kt * 32 + ((lane >> 4) << 3) + (i & 7) + ((i >> 3) << 4);
            const int n = nt * 16 + (lane & 15);
            wbase[S::woff(L) + idx] =
                (k < K && n < N) ? (__bf16)W[k * N + n] : (__bf16)0.0f;
        }
        const float* B = wp.B[L];
        for (int n = threadIdx.x; n < N16; n += blockDim.x)
            bbase[S::boff(L) + n] = (n < N) ? B[n] : 0.0f;
        load_weights<S, L + 1>(wp, wbase, bbase);
    }
}

// Fully unrolled MLP over one 16-row tile; ping-pongs between in/out buffers.
template <class S, int L, int AM>
__device__ __forceinline__ void mlp_layers(float*& in, float*& out,
                                           const __bf16* wbase, const float* bbase,
                                           int lane) {
    if constexpr (L < S::nl) {
        constexpr int K = S::dim(L), N = S::dim(L + 1);
        constexpr int Kp = S::rup32(K), N16 = S::rup16(N);
        constexpr int KT = Kp / 32, NT = N16 / 16;
        constexpr int STRIDE = S::stride();
        constexpr bool ACT = ((AM >> L) & 1) != 0;
        const __bf16* w  = wbase + S::woff(L);
        const float*  bi = bbase + S::boff(L);
#pragma unroll
        for (int nt = 0; nt < NT; ++nt) {
            const int nc = nt * 16 + (lane & 15);
            const float bv = bi[nc];                 // zero-padded bias
            f32x8 c;
#pragma unroll
            for (int r = 0; r < 8; ++r) c[r] = bv;
#pragma unroll
            for (int kt = 0; kt < KT; ++kt) {
                bf16x16 a = afrag<STRIDE>(in, kt * 32, lane);
                bf16x16 b = bfrag<NT>(w, kt, nt, lane);
                c = __builtin_amdgcn_wmma_f32_16x16x32_bf16(
                        false, a, false, b, (short)0, c, false, false);
            }
            const int mb = (lane >> 4) << 3;         // D: rows mb..mb+7, column nc
            if (nc < N) {
#pragma unroll
                for (int r = 0; r < 8; ++r) {
                    float v = c[r];
                    if constexpr (ACT) v = v > 0.0f ? v : 0.2f * v;   // leaky relu
                    out[(mb + r) * STRIDE + nc] = v;
                }
            }
        }
        float* t = in; in = out; out = t;
        mlp_layers<S, L + 1, AM>(in, out, wbase, bbase, lane);
    }
}

template <int EPI, int NSRC, class S, int AM>
__global__ __launch_bounds__(128)
void k_mlp(GatherSrc g0, GatherSrc g1, GatherSrc g2, GatherSrc g3, GatherSrc g4,
           MlpW wp, int nrows, EpiP ep) {
    extern __shared__ char lds_raw[];
    constexpr int NL = S::nl;
    constexpr int STRIDE = S::stride();

    // ---- LDS layout: bf16 weights | f32 biases | per-wave ping-pong tiles ----
    __bf16* wbase = (__bf16*)lds_raw;
    float*  bbase = (float*)(lds_raw + ((S::woff(NL) * 2 + 15) & ~15));
    float*  bufs  = bbase + S::boff(NL);

    load_weights<S, 0>(wp, wbase, bbase);
    __syncthreads();

    const int wave = threadIdx.x >> 5;
    const int lane = threadIdx.x & 31;
    const int row0 = (blockIdx.x * NWAVES + wave) * 16;
    if (row0 >= nrows) return;

    float* bufA = bufs + wave * (2 * 16 * STRIDE);
    float* bufB = bufA + 16 * STRIDE;
    for (int i = lane; i < 16 * STRIDE; i += 32) { bufA[i] = 0.0f; bufB[i] = 0.0f; }

    // ---- gather + concat the sources into the tile ----
    GatherSrc gs[5] = {g0, g1, g2, g3, g4};
    int col = 0;
#pragma unroll
    for (int s = 0; s < NSRC; ++s) {
        const GatherSrc g = gs[s];
        const int tot = 16 * g.width;
        for (int i = lane; i < tot; i += 32) {
            const int r = i / g.width, c = i - r * g.width;
            int row = row0 + r; if (row >= nrows) row = nrows - 1;
            long gi = g.idx ? (long)g.idx[row] : (long)row;
            gi >>= g.shift;
            bufA[r * STRIDE + col + c] = g.ptr[gi * (long)g.rowStride + g.colOff + c];
        }
        col += g.width;
    }

    float* in = bufA; float* out = bufB;
    mlp_layers<S, 0, AM>(in, out, wbase, bbase, lane);
    float* res = in;

    // ---- epilogues ----
    if constexpr (EPI == EPI_ADD) {
        const int N = ep.n;
        for (int i = lane; i < 16 * N; i += 32) {
            const int r = i / N, c = i - r * N, row = row0 + r;
            if (row < nrows) {
                const int gd = ep.sidx[row];
                atomicAdd(&ep.o0[(long)gd * N + c], res[r * STRIDE + c]);
            }
        }
    } else if constexpr (EPI == EPI_I2O_NC) {
        const int N = ep.n;   // 17
        for (int i = lane; i < 16 * N; i += 32) {
            const int r = i / N, c = i - r * N, row = row0 + r;
            if (row < nrows) {
                const int gd = ep.sidx[row];
                const float k = sigmoidf_(res[r * STRIDE]);
                const float v = res[r * STRIDE + c];
                if (c == 0)      atomicAdd(&ep.o2[gd], 1.0f);
                else if (c < 9)  atomicAdd(&ep.o0[gd * 8 + (c - 1)], v * k);
                else             atomicAdd(&ep.o1[gd * 8 + (c - 9)], v * k);
            }
        }
    } else if constexpr (EPI == EPI_I2O_SWI) {
        const int N = ep.n;   // 25
        for (int i = lane; i < 16 * N; i += 32) {
            const int r = i / N, c = i - r * N, row = row0 + r;
            if (row < nrows) {
                const int gd = ep.sidx[row];
                const float k = sigmoidf_(res[r * STRIDE]);
                const float v = res[r * STRIDE + c];
                if (c == 0)      atomicAdd(&ep.o2[gd], 1.0f);
                else if (c < 9)  atomicAdd(&ep.o0[gd * 8 + (c - 1)], v * k * ep.a0[row]);
                else             atomicAdd(&ep.o1[gd * 16 + (c - 9)], v * k);
            }
        }
    } else if constexpr (EPI == EPI_STORE) {
        const int N = ep.n;
        for (int i = lane; i < 16 * N; i += 32) {
            const int r = i / N, c = i - r * N, row = row0 + r;
            if (row < nrows) ep.o0[(long)row * N + c] = res[r * STRIDE + c];
        }
    } else if constexpr (EPI == EPI_SELECT) {
        for (int i = lane; i < 16 * 24; i += 32) {
            const int r = i / 24, c = i - r * 24, row = row0 + r;
            if (row < nrows) {
                const float cv = ep.a1[row];
                ep.o0[(long)row * 24 + c] =
                    (cv > 0.0f) ? res[r * STRIDE + c] : ep.a0[(long)row * 24 + c];
            }
        }
    } else if constexpr (EPI == EPI_LUT) {
        if (lane < 16) {
            const int r = lane, row = row0 + r;
            if (row < nrows) {
                const long e = (long)(row >> 1);
                const float* tb = ep.tab + e * ep.ts + ep.to;   // 64-entry LUT
                float accS = 0.0f, accT = 0.0f;
                for (int i = 0; i < 8; ++i) {
                    const float ai = res[r * STRIDE + i];
                    for (int j = 0; j < 8; ++j) {
                        const float v = sigmoidf_(ai * res[r * STRIDE + 8 + j]);
                        accS += v;
                        accT += tb[i * 8 + j] * v;
                    }
                }
                ep.o0[row] = accT / (accS + 1e-10f);
            }
        }
    } else if constexpr (EPI == EPI_LUTIN) {
        if (lane < 16) {
            const int r = lane, row = row0 + r;
            if (row < nrows) {
                const long e = (long)(row >> 1);
                const float* tb = ep.tab + e * ep.ts + ep.to;   // 8-entry LUT
                float accS = 0.0f, accT = 0.0f;
                for (int j = 0; j < 8; ++j) {
                    const float v = sigmoidf_(res[r * STRIDE + j]);
                    accS += v;
                    accT += tb[j] * v;
                }
                ep.o0[row] = accT / (accS + 1e-10f);
            }
        }
    } else if constexpr (EPI == EPI_FSCALE) {
        if (lane < 16) {
            const int r = lane, row = row0 + r;
            if (row < nrows) {
                const float s0 = sigmoidf_(res[r * STRIDE + 0]);
                const float s1 = sigmoidf_(res[r * STRIDE + 1]);
                const float r0 = ep.a0[(long)row * 2], r1 = ep.a0[(long)row * 2 + 1];
                const float q0 = ep.a1[(long)row * 2], q1 = ep.a1[(long)row * 2 + 1];
                ep.o0[row] = 0.5f * (r0 + r1) * (0.9f + s0 * 0.2f);
                ep.o1[row] = 0.5f * (q0 + q1) * (0.9f + s1 * 0.2f);
            }
        }
    }
}

// ---- small helper kernels --------------------------------------------------

__global__ void k_zero(float* p, long n) {
    const long i = (long)blockIdx.x * blockDim.x + threadIdx.x;
    if (i < n) p[i] = 0.0f;
}

__global__ void k_nc1_norm(float* nfo2, const float* cnt, int N) {
    const int i = blockIdx.x * blockDim.x + threadIdx.x;
    if (i >= N * 8) return;
    float c = cnt[i / 8];
    c = c > 1.0f ? c : 1.0f;
    nfo2[i] /= c;
}

// einsum('eik,ejk->e') = dot_k(colsum_i nm, colsum_j em); then sigmoid, seg-sum
__global__ void k_att_raw(const float* nm, const float* em, const int* nidst,
                          float* att, float* attsum, int E) {
    const int e = blockIdx.x * blockDim.x + threadIdx.x;
    if (e >= E) return;
    const int d = nidst[e];
    float s1[4] = {0, 0, 0, 0}, s2[4] = {0, 0, 0, 0};
#pragma unroll
    for (int i = 0; i < 4; ++i)
#pragma unroll
        for (int k = 0; k < 4; ++k) {
            s1[k] += nm[(long)d * 16 + i * 4 + k];
            s2[k] += em[(long)e * 16 + i * 4 + k];
        }
    const float dot = s1[0] * s2[0] + s1[1] * s2[1] + s1[2] * s2[2] + s1[3] * s2[3];
    const float raw = 1.0f / (1.0f + __expf(-dot));
    att[e] = raw;
    atomicAdd(&attsum[d], raw);
}

__global__ void k_att_norm(float* att, const float* attsum, const int* nidst, int E) {
    const int e = blockIdx.x * blockDim.x + threadIdx.x;
    if (e >= E) return;
    att[e] = att[e] / (attsum[nidst[e]] + 0.001f);
}

// ---- host side -------------------------------------------------------------

struct PW { const float* W; const float* B; };

static GatherSrc gsrc(const float* p, const int* idx, int rowStride, int colOff,
                      int width, int shift = 0) {
    GatherSrc g; g.ptr = p; g.idx = idx; g.rowStride = rowStride;
    g.colOff = colOff; g.width = width; g.shift = shift;
    return g;
}

template <int EPI, int NSRC, int NL, int D0, int D1, int D2, int D3, int D4, int D5, int AM>
static void launch_mlp(hipStream_t st, GatherSrc g0, GatherSrc g1, GatherSrc g2,
                       GatherSrc g3, GatherSrc g4, const PW* p, long nrows,
                       const EpiP& ep) {
    using S = Shape<NL, D0, D1, D2, D3, D4, D5>;
    MlpW w{};
    for (int l = 0; l < NL; ++l) { w.W[l] = p[l].W; w.B[l] = p[l].B; }
    const long tiles = (nrows + 15) / 16;
    const int  grd   = (int)((tiles + NWAVES - 1) / NWAVES);
    k_mlp<EPI, NSRC, S, AM><<<grd, 128, S::smem(), st>>>(g0, g1, g2, g3, g4, w,
                                                         (int)nrows, ep);
}

static void zero_buf(hipStream_t st, float* p, long n) {
    k_zero<<<(int)((n + 255) / 256), 256, 0, st>>>(p, n);
}

extern "C" void kernel_launch(void* const* d_in, const int* in_sizes, int n_in,
                              void* d_out, int out_size, void* d_ws, size_t ws_size,
                              hipStream_t stream) {
    const float* nf     = (const float*)d_in[0];
    const float* onehot = (const float*)d_in[1];
    const float* nefo   = (const float*)d_in[2];
    const float* nefi   = (const float*)d_in[3];
    const float* ef     = (const float*)d_in[4];
    const float* efin   = (const float*)d_in[5];
    const int* no_src = (const int*)d_in[6];
    const int* no_dst = (const int*)d_in[7];
    const int* ni_src = (const int*)d_in[8];
    const int* ni_dst = (const int*)d_in[9];
    const int* ce_src = (const int*)d_in[10];
    const int* ce_dst = (const int*)d_in[11];
    const float* P = (const float*)d_in[14];   // flattened params pytree

    const int N  = in_sizes[0] / 24;
    const int E  = in_sizes[2] / 13;
    const int EC = in_sizes[4] / 81;

    // ---- workspace layout (floats) ----
    float* Wp      = (float*)d_ws;
    float* new_nf  = Wp;                          // 24N  --+
    float* nfo1    = new_nf + (long)24 * N;       // 8N     | contiguous 50N
    float* nfo2    = nfo1   + (long)8 * N;        // 16N    | zeroed per conv
    float* cnt     = nfo2   + (long)16 * N;       // N      |
    float* attsum  = cnt    + N;                  // N   --+
    float* hA      = attsum + N;                  // 24N
    float* hB      = hA     + (long)24 * N;       // 24N
    float* nodemat = hB     + (long)24 * N;       // 16N
    float* edgemat = nodemat + (long)16 * N;      // 16E
    float* att     = edgemat + (long)16 * E;      // E
    float* qbuf    = att     + E;                 // 6EC
    float* r2      = qbuf    + (long)6 * EC;      // 2EC
    float* rin     = r2      + (long)2 * EC;      // 2EC

    // ---- output layout: h | cef | cef_in | dri_sca ----
    float* h_out = (float*)d_out;
    float* cef   = h_out + (long)24 * N;
    float* cefin = cef + EC;
    float* dri   = cefin + EC;

    // ---- walk params (jax pytree: dict keys sorted; lists of (W,b)) ----
    long off = 0;
    auto nx = [&](int k, int n) {
        PW p{P + off, P + off + (long)k * n};
        off += (long)k * n + n;
        return p;
    };
    auto walk = [&](PW* o, const int* d, int nl) {
        for (int l = 0; l < nl; ++l) o[l] = nx(d[l], d[l + 1]);
    };

    int d_fs[]  = {76, 16, 16, 2};
    int d_la[]  = {19, 16, 16, 16};
    int d_li[]  = {10, 16, 16, 8};
    int d_lq[]  = {72, 16, 16, 6};
    int d_dri[] = {4, 2, 1};
    int d_emb[] = {20, 24, 12, 4};
    int d_n1i[] = {61, 16, 17};
    int d_n1o[] = {61, 16, 24};
    int d_n1r[] = {40, 16, 24};
    int d_et[]  = {37, 24, 24, 24, 16};
    int d_si[]  = {61, 24, 24, 24, 25};
    int d_so[]  = {61, 24, 24, 24, 24};
    int d_sr[]  = {48, 24, 24, 24, 24};
    int d_t1[]  = {24, 24, 24, 24, 16};

    PW p_fs[3], p_la[3], p_li[3], p_lq[3], p_dri[2], p_emb[3];
    PW p_n1i[2], p_n1o[2], p_n1r[2];
    PW p2_et[4], p2_si[4], p2_so[4], p2_sr[4], p2_t1[4];
    PW p3_et[4], p3_si[4], p3_so[4], p3_sr[4], p3_t1[4];

    // sorted order: cell{for_scale,lut_att,lut_att_in,lut_query}, dri_mlp,
    // embedding, nc1{msg_i2o,msg_o2i,reduce_o},
    // nc2{edge_trans1,msg_i2o,msg_o2i,reduce_o,trans1}, nc3{...}
    walk(p_fs, d_fs, 3);  walk(p_la, d_la, 3);  walk(p_li, d_li, 3);  walk(p_lq, d_lq, 3);
    walk(p_dri, d_dri, 2); walk(p_emb, d_emb, 3);
    walk(p_n1i, d_n1i, 2); walk(p_n1o, d_n1o, 2); walk(p_n1r, d_n1r, 2);
    walk(p2_et, d_et, 4); walk(p2_si, d_si, 4); walk(p2_so, d_so, 4);
    walk(p2_sr, d_sr, 4); walk(p2_t1, d_t1, 4);
    walk(p3_et, d_et, 4); walk(p3_si, d_si, 4); walk(p3_so, d_so, 4);
    walk(p3_sr, d_sr, 4); walk(p3_t1, d_t1, 4);

    GatherSrc Z{};

    // ---- embedding + dri_mlp fused chain: 20->24->12->4->2->1 ----
    {
        PW p_chain[5] = {p_emb[0], p_emb[1], p_emb[2], p_dri[0], p_dri[1]};
        EpiP ep{}; ep.o0 = dri; ep.n = 1;
        // acts {1,1,0,1,0} -> mask 0b01011 = 11
        launch_mlp<EPI_STORE, 1, 5, 20, 24, 12, 4, 2, 1, 11>(
            stream, gsrc(onehot, nullptr, 20, 0, 20), Z, Z, Z, Z, p_chain, N, ep);
    }

    // ---- nc1 (plain net_conv) ----
    zero_buf(stream, new_nf, (long)50 * N);
    {
        EpiP ep{}; ep.o0 = new_nf; ep.sidx = no_dst; ep.n = 24;
        launch_mlp<EPI_ADD, 3, 2, 61, 16, 24, 0, 0, 0, 1>(
            stream, gsrc(nf, no_src, 24, 0, 24), gsrc(nf, no_dst, 24, 0, 24),
            gsrc(nefo, nullptr, 13, 0, 13), Z, Z, p_n1o, E, ep);
    }
    {
        EpiP ep{}; ep.o0 = nfo1; ep.o1 = nfo2; ep.o2 = cnt; ep.sidx = ni_dst; ep.n = 17;
        launch_mlp<EPI_I2O_NC, 3, 2, 61, 16, 17, 0, 0, 0, 1>(
            stream, gsrc(nf, ni_src, 24, 0, 24), gsrc(nf, ni_dst, 24, 0, 24),
            gsrc(nefi, nullptr, 13, 0, 13), Z, Z, p_n1i, E, ep);
    }
    k_nc1_norm<<<(N * 8 + 255) / 256, 256, 0, stream>>>(nfo2, cnt, N);
    {
        EpiP ep{}; ep.o0 = hA; ep.a0 = new_nf; ep.a1 = cnt; ep.n = 24;
        launch_mlp<EPI_SELECT, 3, 2, 40, 16, 24, 0, 0, 0, 1>(
            stream, gsrc(new_nf, nullptr, 24, 0, 24), gsrc(nfo1, nullptr, 8, 0, 8),
            gsrc(nfo2, nullptr, 8, 0, 8), Z, Z, p_n1r, N, ep);
    }

    // ---- nc2 / nc3 (net_conv_swi): same shapes -> shared instantiations ----
    auto swi = [&](PW* et, PW* si, PW* so, PW* sr, PW* t1,
                   const float* hin, float* hout) {
        zero_buf(stream, new_nf, (long)50 * N);
        {
            EpiP ep{}; ep.o0 = new_nf; ep.sidx = no_dst; ep.n = 24;
            launch_mlp<EPI_ADD, 3, 4, 61, 24, 24, 24, 24, 0, 7>(
                stream, gsrc(hin, no_src, 24, 0, 24), gsrc(hin, no_dst, 24, 0, 24),
                gsrc(nefo, nullptr, 13, 0, 13), Z, Z, so, E, ep);
        }
        {
            EpiP ep{}; ep.o0 = nodemat; ep.n = 16;
            launch_mlp<EPI_STORE, 1, 4, 24, 24, 24, 24, 16, 0, 7>(
                stream, gsrc(new_nf, nullptr, 24, 0, 24), Z, Z, Z, Z, t1, N, ep);
        }
        {
            EpiP ep{}; ep.o0 = edgemat; ep.n = 16;
            launch_mlp<EPI_STORE, 2, 4, 37, 24, 24, 24, 16, 0, 7>(
                stream, gsrc(new_nf, ni_src, 24, 0, 24),
                gsrc(nefi, nullptr, 13, 0, 13), Z, Z, Z, et, E, ep);
        }
        k_att_raw<<<(E + 255) / 256, 256, 0, stream>>>(nodemat, edgemat, ni_dst,
                                                       att, attsum, E);
        k_att_norm<<<(E + 255) / 256, 256, 0, stream>>>(att, attsum, ni_dst, E);
        {
            EpiP ep{}; ep.o0 = nfo1; ep.o1 = nfo2; ep.o2 = cnt; ep.sidx = ni_dst;
            ep.a0 = att; ep.n = 25;
            launch_mlp<EPI_I2O_SWI, 3, 4, 61, 24, 24, 24, 25, 0, 7>(
                stream, gsrc(new_nf, ni_src, 24, 0, 24), gsrc(hin, ni_dst, 24, 0, 24),
                gsrc(nefi, nullptr, 13, 0, 13), Z, Z, si, E, ep);
        }
        {
            EpiP ep{}; ep.o0 = hout; ep.a0 = new_nf; ep.a1 = cnt; ep.n = 24;
            launch_mlp<EPI_SELECT, 3, 4, 48, 24, 24, 24, 24, 0, 7>(
                stream, gsrc(new_nf, nullptr, 24, 0, 24), gsrc(nfo1, nullptr, 8, 0, 8),
                gsrc(nfo2, nullptr, 16, 0, 16), Z, Z, sr, N, ep);
        }
    };
    swi(p2_et, p2_si, p2_so, p2_sr, p2_t1, hA, hB);
    swi(p3_et, p3_si, p3_so, p3_sr, p3_t1, hB, h_out);

    // ---- cell_conv ----
    {
        EpiP ep{}; ep.o0 = qbuf; ep.n = 6;
        launch_mlp<EPI_STORE, 3, 3, 72, 16, 16, 6, 0, 0, 3>(
            stream, gsrc(h_out, ce_src, 24, 0, 24), gsrc(nf, ce_src, 24, 0, 24),
            gsrc(nf, ce_dst, 24, 0, 24), Z, Z, p_lq, EC, ep);
    }
    {   // lut_att over (e,d) rows: input = [q2(2) | ef[:, :17] repeated by D]
        EpiP ep{}; ep.o0 = r2; ep.tab = ef; ep.ts = 81; ep.to = 17;
        launch_mlp<EPI_LUT, 2, 3, 19, 16, 16, 16, 0, 0, 3>(
            stream, gsrc(qbuf, nullptr, 3, 0, 2), gsrc(ef, nullptr, 81, 0, 17, 1),
            Z, Z, Z, p_la, (long)2 * EC, ep);
    }
    {   // lut_att_in: input = [q1(1) | ef_input[:, :9] repeated by D]
        EpiP ep{}; ep.o0 = rin; ep.tab = efin; ep.ts = 17; ep.to = 9;
        launch_mlp<EPI_LUTIN, 2, 3, 10, 16, 16, 8, 0, 0, 3>(
            stream, gsrc(qbuf, nullptr, 3, 2, 1), gsrc(efin, nullptr, 17, 0, 9, 1),
            Z, Z, Z, p_li, (long)2 * EC, ep);
    }
    {
        EpiP ep{}; ep.o0 = cef; ep.o1 = cefin; ep.a0 = r2; ep.a1 = rin; ep.n = 2;
        launch_mlp<EPI_FSCALE, 5, 3, 76, 16, 16, 2, 0, 0, 3>(
            stream, gsrc(h_out, ce_src, 24, 0, 24), gsrc(nf, ce_src, 24, 0, 24),
            gsrc(nf, ce_dst, 24, 0, 24), gsrc(r2, nullptr, 2, 0, 2),
            gsrc(rin, nullptr, 2, 0, 2), p_fs, EC, ep);
    }
    (void)n_in; (void)out_size; (void)ws_size; (void)in_sizes; (void)ce_dst;
}
